// SemanticKnowledgeGraph_55722905699140
// MI455X (gfx1250) — compile-verified, matrix operand order checked by reading, not run on hardware
//
#include <hip/hip_runtime.h>

#define N_NODES 16384
#define N_EDGES 262144LL
#define DN 256
#define DE 64
#define NH 8
#define EPS_LN 1e-5f

typedef __attribute__((ext_vector_type(16))) __bf16 v16bf;
typedef __attribute__((ext_vector_type(8)))  float  v8f;

union ABf16 { v16bf v; uint4 q[2]; __bf16 e[16]; };

__device__ __forceinline__ float leaky(float x) { return x > 0.f ? x : 0.2f * x; }

__device__ __forceinline__ unsigned f2ord(float f) {
    unsigned u = __float_as_uint(f);
    return (u & 0x80000000u) ? ~u : (u | 0x80000000u);
}
__device__ __forceinline__ float ord2f(unsigned o) {
    return (o & 0x80000000u) ? __uint_as_float(o & 0x7fffffffu) : __uint_as_float(~o);
}

// ---------------------------------------------------------------------------
// CDNA5 async global->LDS copy machinery (ASYNCcnt-tracked, ISA ch.10/15.18).
// Inline asm: robust across the two toolchains' differing builtin arities.
// ---------------------------------------------------------------------------
typedef __attribute__((address_space(3))) const void lds_cv;

__device__ __forceinline__ unsigned lds_off_of(const void* p) {
    return (unsigned)(size_t)(lds_cv*)p;   // AS3 pointer value == LDS byte offset
}

__device__ __forceinline__ void async_ld16(unsigned lds_off, const void* g) {
    asm volatile("global_load_async_to_lds_b128 %0, %1, off"
                 :: "v"(lds_off), "v"((unsigned long long)(size_t)g)
                 : "memory");
}

__device__ __forceinline__ void wait_async_zero() {
    asm volatile("s_wait_asynccnt 0x0" ::: "memory");
}

// whole-block cooperative copy, 16B per lane per step
__device__ __forceinline__ void copy_panel(const __bf16* g, __bf16* l, int bytes) {
    const int tid = threadIdx.x, nthr = blockDim.x;
    const unsigned lbase = lds_off_of(l);
    const char* gb = (const char*)g;
    for (int o = tid * 16; o < bytes; o += nthr * 16)
        async_ld16(lbase + o, gb + o);
}

// ---------------------------------------------------------------------------
// WMMA fragment loaders.
// 16-bit A fragment (16x32 MxK), wave32: lane L holds row m=L&15, half=L>>4.
// elements 0..7  -> K = kbase + half*8 + e        (8 contiguous bf16 = b128)
// elements 8..15 -> K = kbase + 16 + half*8 + e-8 (8 contiguous bf16 = b128)
// Callers pre-fold  A + row*ld + half*8  into `p`; kbase stays a literal so
// loads become immediate-offset b128 loads. These helpers are force-inlined,
// so the address space of the argument (LDS shared array vs global) is
// preserved and LDS accesses lower to ds_load_b128.
// ---------------------------------------------------------------------------
__device__ __forceinline__ v16bf load_a_frag(const __bf16* p) {
    ABf16 u;
    u.q[0] = *reinterpret_cast<const uint4*>(p);
    u.q[1] = *reinterpret_cast<const uint4*>(p + 16);
    return u.v;
}

__device__ __forceinline__ v16bf load_a_frag_f32(const float* p) {
    ABf16 u;
#pragma unroll
    for (int i = 0; i < 8; ++i) u.e[i] = (__bf16)p[i];
#pragma unroll
    for (int i = 0; i < 8; ++i) u.e[8 + i] = (__bf16)p[16 + i];
    return u.v;
}

// B fragment from packed layout: a 16x32 tile is a contiguous 512-element
// block; lane L owns elements [L*16, L*16+16) -> two b128 loads.
__device__ __forceinline__ v16bf load_b_frag(const __bf16* tile) {
    const int lane = threadIdx.x & 31;
    const __bf16* p = tile + lane * 16;
    ABf16 u;
    u.q[0] = reinterpret_cast<const uint4*>(p)[0];
    u.q[1] = reinterpret_cast<const uint4*>(p)[1];
    return u.v;
}

#define WMMA_BF16(a, b, c) \
    __builtin_amdgcn_wmma_f32_16x16x32_bf16(false, (a), false, (b), (short)0, (c), false, false)

// ---------------------------------------------------------------------------
// Repack a row-major [K,N] matrix (f32 or bf16 source, zero padding past
// Ksrc/Nsrc) into the fragment-linear WMMA-B layout.
// ---------------------------------------------------------------------------
template <typename T>
__global__ void repack_b_kernel(const T* __restrict__ src, __bf16* __restrict__ dst,
                                int K, int N, int ld, int Ksrc, int Nsrc) {
    long long id = (long long)blockIdx.x * blockDim.x + threadIdx.x;
    if (id >= (long long)K * N) return;
    int blk = (int)(id >> 9);
    int r = (int)(id & 511);
    int lane = r >> 4, e = r & 15;
    int ktiles = K >> 5;
    int kt = blk % ktiles, nt = blk / ktiles;
    int half = lane >> 4;
    int n = nt * 16 + (lane & 15);
    int kk = (e < 8) ? (half * 8 + e) : (16 + half * 8 + (e - 8));
    int k = kt * 32 + kk;
    float v = 0.f;
    if (k < Ksrc && n < Nsrc) v = (float)src[(long long)k * ld + n];
    dst[id] = (__bf16)v;
}

__global__ void cvt_bf16_kernel(const float* __restrict__ src, __bf16* __restrict__ dst, long long n) {
    long long i = (long long)blockIdx.x * blockDim.x + threadIdx.x;
    if (i < n) dst[i] = (__bf16)src[i];
}

// c[j] = a1[j] + sum_i bq[i]*A1[i,j] + bk[i]*A1[256+i,j] + be[i]*A1[512+i,j]
__global__ void cvec_kernel(const float* __restrict__ A1, const float* __restrict__ a1,
                            const float* __restrict__ bq, const float* __restrict__ bk,
                            const float* __restrict__ be, float* __restrict__ cv) {
    int j = threadIdx.x;
    float acc = a1[j];
    for (int i = 0; i < DN; ++i) {
        acc += bq[i] * A1[(long long)i * DN + j];
        acc += bk[i] * A1[(long long)(DN + i) * DN + j];
        acc += be[i] * A1[(long long)(2 * DN + i) * DN + j];
    }
    cv[j] = acc;
}

// ---------------------------------------------------------------------------
// Wave-per-tile WMMA GEMMs; block = 256 threads = 8 waves = 8 consecutive
// m-tiles sharing one nt. When Mtiles%8==0 the shared B panel (Ktiles tiles)
// is async-copied into LDS once per block and read back as ds_load_b128.
// The staged / unstaged K-loops are kept in separate branches so the B
// pointer never becomes a generic (flat) pointer.
// Grids must launch exactly Mtiles*Ntiles tiles.
// ---------------------------------------------------------------------------
__global__ void gemm_bf16_f32(const __bf16* __restrict__ A, int lda,
                              const __bf16* __restrict__ Bpk,
                              const float* __restrict__ bias,
                              float* __restrict__ C, int ldc,
                              int Mtiles, int Ntiles, int Ktiles, int K) {
    __shared__ __align__(16) __bf16 bpan[4096];  // up to 8 B tiles (8 KB)
    const int wave = threadIdx.x >> 5, lane = threadIdx.x & 31;
    const int half = lane >> 4;
    const int tile = blockIdx.x * (blockDim.x >> 5) + wave;
    const int mt = tile % Mtiles, nt = tile / Mtiles;
    const bool staged = (Mtiles % (blockDim.x >> 5)) == 0;  // grid-uniform

    const __bf16* ap = A + ((size_t)mt * 16 + (lane & 15)) * lda + half * 8;
    v8f acc = {};
    if (staged) {
        copy_panel(Bpk + (size_t)nt * Ktiles * 512, bpan, Ktiles * 1024);
        wait_async_zero();
        __syncthreads();
        for (int kt = 0; kt < Ktiles; ++kt) {
            v16bf a = load_a_frag(ap + kt * 32);
            v16bf b = load_b_frag(bpan + kt * 512);          // AS3 -> ds_load_b128
            acc = WMMA_BF16(a, b, acc);
        }
    } else {
        const __bf16* bp = Bpk + (size_t)nt * Ktiles * 512;  // AS1 -> global_load
        for (int kt = 0; kt < Ktiles; ++kt) {
            v16bf a = load_a_frag(ap + kt * 32);
            v16bf b = load_b_frag(bp + kt * 512);
            acc = WMMA_BF16(a, b, acc);
        }
    }
    const int n = nt * 16 + (lane & 15);
    const float bb = bias ? bias[n] : 0.f;
#pragma unroll
    for (int r = 0; r < 8; ++r) {
        int m = mt * 16 + half * 8 + r;
        C[(long long)m * ldc + n] = acc[r] + bb;
    }
}

__global__ void gemm_f32_bf16(const float* __restrict__ A, int lda,
                              const __bf16* __restrict__ Bpk,
                              __bf16* __restrict__ C, int ldc,
                              int Mtiles, int Ntiles, int Ktiles, int K) {
    __shared__ __align__(16) __bf16 bpan[4096];
    const int wave = threadIdx.x >> 5, lane = threadIdx.x & 31;
    const int half = lane >> 4;
    const int tile = blockIdx.x * (blockDim.x >> 5) + wave;
    const int mt = tile % Mtiles, nt = tile / Mtiles;
    const bool staged = (Mtiles % (blockDim.x >> 5)) == 0;

    const float* ap = A + ((size_t)mt * 16 + (lane & 15)) * lda + half * 8;
    v8f acc = {};
    if (staged) {
        copy_panel(Bpk + (size_t)nt * Ktiles * 512, bpan, Ktiles * 1024);
        wait_async_zero();
        __syncthreads();
        for (int kt = 0; kt < Ktiles; ++kt) {
            v16bf a = load_a_frag_f32(ap + kt * 32);
            v16bf b = load_b_frag(bpan + kt * 512);
            acc = WMMA_BF16(a, b, acc);
        }
    } else {
        const __bf16* bp = Bpk + (size_t)nt * Ktiles * 512;
        for (int kt = 0; kt < Ktiles; ++kt) {
            v16bf a = load_a_frag_f32(ap + kt * 32);
            v16bf b = load_b_frag(bp + kt * 512);
            acc = WMMA_BF16(a, b, acc);
        }
    }
    const int n = nt * 16 + (lane & 15);
#pragma unroll
    for (int r = 0; r < 8; ++r) {
        int m = mt * 16 + half * 8 + r;
        C[(long long)m * ldc + n] = (__bf16)acc[r];
    }
}

// ---------------------------------------------------------------------------
// Fused per-edge attention MLP. One wave = 16 edges; 4 waves per block.
// Gathered A fragments live in VGPRs. The P/Q/R fragment panel for each nt
// (identical across waves) is async-copied into LDS with double buffering so
// the copy of panel nt+1 overlaps the 18 WMMAs of panel nt. hidden flips
// C-layout -> A-layout through LDS for the second WMMA chain (@A2, N padded
// to 16), then head-mean via __shfl_xor and an ordered-uint global atomicMax.
// ---------------------------------------------------------------------------
__global__ __launch_bounds__(128, 1) void edge_attn_kernel(
    const __bf16* __restrict__ nfb, const long long* __restrict__ ei,
    const float* __restrict__ ef,
    const __bf16* __restrict__ Ppk, const __bf16* __restrict__ Qpk,
    const __bf16* __restrict__ Rpk, const float* __restrict__ cv,
    const __bf16* __restrict__ A2pk, const float* __restrict__ a2,
    float* __restrict__ s_out, unsigned* __restrict__ maxord, long long E) {
    __shared__ __align__(16) __bf16 bpan[2][9216];   // P(8)|Q(8)|R(2) tiles, 18 KB x2
    __shared__ __align__(16) __bf16 hid[4][16 * DN]; // 8 KB per wave
    const int wave = threadIdx.x >> 5, lane = threadIdx.x & 31;
    const int m = lane & 15, half = lane >> 4;
    const long long e0 = ((long long)blockIdx.x * 4 + wave) * 16;
    const long long srow = ei[e0 + m];       // src node of edge (e0+m)
    const long long trow = ei[E + e0 + m];   // tgt node

    const __bf16* sp = nfb + srow * DN + half * 8;
    const __bf16* tp = nfb + trow * DN + half * 8;
    const float*  fp = ef + (e0 + m) * DE + half * 8;

    v16bf aq[8], ak[8], ae[2];
#pragma unroll
    for (int kt = 0; kt < 8; ++kt) aq[kt] = load_a_frag(sp + kt * 32);
#pragma unroll
    for (int kt = 0; kt < 8; ++kt) ak[kt] = load_a_frag(tp + kt * 32);
#pragma unroll
    for (int kt = 0; kt < 2; ++kt) ae[kt] = load_a_frag_f32(fp + kt * 32);

    // prologue: stage panel nt=0 into buffer 0
    copy_panel(Ppk, &bpan[0][0], 8192);
    copy_panel(Qpk, &bpan[0][4096], 8192);
    copy_panel(Rpk, &bpan[0][8192], 2048);

    for (int nt = 0; nt < 16; ++nt) {
        const int cur = nt & 1;
        wait_async_zero();    // own wave's copies into bpan[cur] complete
        __syncthreads();      // everyone's copies complete; prev compute done
        if (nt < 15) {        // overlap next panel copy with this nt's WMMAs
            copy_panel(Ppk + (size_t)(nt + 1) * 8 * 512, &bpan[1 - cur][0], 8192);
            copy_panel(Qpk + (size_t)(nt + 1) * 8 * 512, &bpan[1 - cur][4096], 8192);
            copy_panel(Rpk + (size_t)(nt + 1) * 2 * 512, &bpan[1 - cur][8192], 2048);
        }
        v8f acc = {};
#pragma unroll
        for (int kt = 0; kt < 8; ++kt)
            acc = WMMA_BF16(aq[kt], load_b_frag(&bpan[cur][kt * 512]), acc);
#pragma unroll
        for (int kt = 0; kt < 8; ++kt)
            acc = WMMA_BF16(ak[kt], load_b_frag(&bpan[cur][4096 + kt * 512]), acc);
#pragma unroll
        for (int kt = 0; kt < 2; ++kt)
            acc = WMMA_BF16(ae[kt], load_b_frag(&bpan[cur][8192 + kt * 512]), acc);
        const int n = nt * 16 + m;
        const float cb = cv[n];
#pragma unroll
        for (int r = 0; r < 8; ++r) {
            float x = leaky(acc[r] + cb);
            hid[wave][(half * 8 + r) * DN + n] = (__bf16)x;
        }
    }
    __syncthreads();

    v8f acc2 = {};
    const __bf16* hp = &hid[wave][m * DN + half * 8];
#pragma unroll
    for (int kt = 0; kt < 8; ++kt)
        acc2 = WMMA_BF16(load_a_frag(hp + kt * 32), load_b_frag(A2pk + kt * 512), acc2);

    // scores: column n = lane&15 (valid for n < NH), rows = half*8 + r
    float sc[8];
#pragma unroll
    for (int r = 0; r < 8; ++r)
        sc[r] = (m < NH) ? leaky(acc2[r] + a2[m]) : 0.f;
#pragma unroll
    for (int mask = 1; mask < 8; mask <<= 1)
#pragma unroll
        for (int r = 0; r < 8; ++r) sc[r] += __shfl_xor(sc[r], mask, 32);

    if ((lane & 15) == 0) {  // lanes 0 (rows 0..7) and 16 (rows 8..15)
        float lm = -3.4e38f;
#pragma unroll
        for (int r = 0; r < 8; ++r) {
            float sv = sc[r] * (1.0f / NH);
            s_out[e0 + half * 8 + r] = sv;
            lm = fmaxf(lm, sv);
        }
        atomicMax(maxord, f2ord(lm));
    }
}

__global__ void softmax_exp_kernel(float* __restrict__ s, const unsigned* __restrict__ maxord,
                                   float* __restrict__ gsum, long long E) {
    __shared__ float red[256];
    long long i = (long long)blockIdx.x * 256 + threadIdx.x;
    float mx = ord2f(*maxord);
    float t = 0.f;
    if (i < E) { t = __expf(s[i] - mx); s[i] = t; }
    red[threadIdx.x] = t;
    __syncthreads();
    for (int st = 128; st > 0; st >>= 1) {
        if ((int)threadIdx.x < st) red[threadIdx.x] += red[threadIdx.x + st];
        __syncthreads();
    }
    if (threadIdx.x == 0) atomicAdd(gsum, red[0]);
}

// out[src] += w_e * v[tgt] ; 64 lanes x 4 floats per edge, 4 edges per block
__global__ void scatter_kernel(const float* __restrict__ s, const float* __restrict__ gsum,
                               const long long* __restrict__ ei, const float* __restrict__ v,
                               float* __restrict__ oacc, long long E) {
    int t = threadIdx.x;
    long long e = (long long)blockIdx.x * 4 + (t >> 6);
    int c = (t & 63) * 4;
    float w = s[e] / *gsum;
    long long srcn = ei[e], tgtn = ei[E + e];
    const float* vp = v + tgtn * DN + c;
    float* op = oacc + srcn * DN + c;
#pragma unroll
    for (int i = 0; i < 4; ++i) atomicAdd(op + i, w * vp[i]);
}

__global__ void ln_kernel(const float* __restrict__ h, const float* __restrict__ oacc,
                          const float* __restrict__ gamma, const float* __restrict__ beta,
                          float* __restrict__ out) {
    __shared__ float red[256];
    int row = blockIdx.x, t = threadIdx.x;
    long long idx = (long long)row * DN + t;
    float x = h[idx] + oacc[idx];
    red[t] = x;
    __syncthreads();
    for (int st = 128; st > 0; st >>= 1) {
        if (t < st) red[t] += red[t + st];
        __syncthreads();
    }
    float mu = red[0] * (1.f / DN);
    __syncthreads();
    float d = x - mu;
    red[t] = d * d;
    __syncthreads();
    for (int st = 128; st > 0; st >>= 1) {
        if (t < st) red[t] += red[t + st];
        __syncthreads();
    }
    float var = red[0] * (1.f / DN);
    out[idx] = d * rsqrtf(var + EPS_LN) * gamma[t] + beta[t];
}

// ---------------------------------------------------------------------------
extern "C" void kernel_launch(void* const* d_in, const int* in_sizes, int n_in,
                              void* d_out, int out_size, void* d_ws, size_t ws_size,
                              hipStream_t stream) {
    (void)in_sizes; (void)n_in; (void)out_size; (void)ws_size;
    const float* nf = (const float*)d_in[0];
    const long long* ei = (const long long*)d_in[1];
    const float* ef = (const float*)d_in[2];
    const float* Wn = (const float*)d_in[3];  const float* bn = (const float*)d_in[4];
    const float* Wq = (const float*)d_in[5];  const float* bq = (const float*)d_in[6];
    const float* Wk = (const float*)d_in[7];  const float* bk = (const float*)d_in[8];
    const float* Wv = (const float*)d_in[9];  const float* bv = (const float*)d_in[10];
    const float* We = (const float*)d_in[11]; const float* be = (const float*)d_in[12];
    const float* A1 = (const float*)d_in[13]; const float* a1 = (const float*)d_in[14];
    const float* A2 = (const float*)d_in[15]; const float* a2 = (const float*)d_in[16];
    const float* gamma = (const float*)d_in[17]; const float* beta = (const float*)d_in[18];
    float* out = (float*)d_out;

    char* w = (char*)d_ws;
    size_t off = 0;
    auto alloc = [&](size_t bytes) -> char* {
        char* p = w + off;
        off = (off + bytes + 255) & ~(size_t)255;
        return p;
    };
    __bf16* nfb   = (__bf16*)alloc((size_t)N_NODES * DN * 2);
    __bf16* WnPk  = (__bf16*)alloc(DN * DN * 2);
    __bf16* WvPk  = (__bf16*)alloc(DN * DN * 2);
    __bf16* A1aPk = (__bf16*)alloc(DN * DN * 2);
    __bf16* A1bPk = (__bf16*)alloc(DN * DN * 2);
    __bf16* A1cPk = (__bf16*)alloc(DN * DN * 2);
    __bf16* Prm   = (__bf16*)alloc(DN * DN * 2);
    __bf16* Qrm   = (__bf16*)alloc(DN * DN * 2);
    __bf16* Rrm   = (__bf16*)alloc(DE * DN * 2);
    __bf16* Ppk   = (__bf16*)alloc(DN * DN * 2);
    __bf16* Qpk   = (__bf16*)alloc(DN * DN * 2);
    __bf16* Rpk   = (__bf16*)alloc(DE * DN * 2);
    __bf16* A2pk  = (__bf16*)alloc(DN * 16 * 2);
    float*  cv    = (float*)alloc(DN * 4);
    float*  h_ws  = (float*)alloc((size_t)N_NODES * DN * 4);
    float*  v_ws  = (float*)alloc((size_t)N_NODES * DN * 4);
    float*  oacc  = (float*)alloc((size_t)N_NODES * DN * 4);
    float*  s_ws  = (float*)alloc((size_t)N_EDGES * 4);
    unsigned* maxord = (unsigned*)alloc(4);
    float*  gsum  = (float*)alloc(4);

    hipMemsetAsync(oacc, 0, (size_t)N_NODES * DN * 4, stream);
    hipMemsetAsync(maxord, 0, 4, stream);  // ordered-uint 0 == most-negative float
    hipMemsetAsync(gsum, 0, 4, stream);

    {   // node features -> bf16 (L2-resident gather source for the edge kernel)
        long long n = (long long)N_NODES * DN;
        cvt_bf16_kernel<<<(int)((n + 255) / 256), 256, 0, stream>>>(nf, nfb, n);
    }

    auto repack_f = [&](const float* src, __bf16* dst, int K, int N, int ld, int Ks, int Ns) {
        long long tot = (long long)K * N;
        repack_b_kernel<float><<<(int)((tot + 255) / 256), 256, 0, stream>>>(src, dst, K, N, ld, Ks, Ns);
    };
    auto repack_h = [&](const __bf16* src, __bf16* dst, int K, int N, int ld, int Ks, int Ns) {
        long long tot = (long long)K * N;
        repack_b_kernel<__bf16><<<(int)((tot + 255) / 256), 256, 0, stream>>>(src, dst, K, N, ld, Ks, Ns);
    };

    repack_f(Wn, WnPk, DN, DN, DN, DN, DN);
    repack_f(Wv, WvPk, DN, DN, DN, DN, DN);
    repack_f(A1,                       A1aPk, DN, DN, DN, DN, DN);
    repack_f(A1 + (size_t)DN * DN,     A1bPk, DN, DN, DN, DN, DN);
    repack_f(A1 + (size_t)2 * DN * DN, A1cPk, DN, DN, DN, DN, DN);
    repack_f(A2, A2pk, DN, 16, NH, DN, NH);  // pad heads 8..15 with zeros

    cvec_kernel<<<1, 256, 0, stream>>>(A1, a1, bq, bk, be, cv);

    // Folded projections: P = Wq@A1[0:256], Q = Wk@A1[256:512], R = We@A1[512:768]
    gemm_f32_bf16<<<32, 256, 0, stream>>>(Wq, DN, A1aPk, Prm, DN, 16, 16, 8, 256);
    gemm_f32_bf16<<<32, 256, 0, stream>>>(Wk, DN, A1bPk, Qrm, DN, 16, 16, 8, 256);
    gemm_f32_bf16<<<8,  256, 0, stream>>>(We, DN, A1cPk, Rrm, DN, 4, 16, 8, 256);
    repack_h(Prm, Ppk, DN, DN, DN, DN, DN);
    repack_h(Qrm, Qpk, DN, DN, DN, DN, DN);
    repack_h(Rrm, Rpk, DE, DN, DN, DE, DN);

    // h = nf@Wn + bn ; v = nf@Wv + bv   (q,k,edge_h folded away)
    gemm_bf16_f32<<<2048, 256, 0, stream>>>(nfb, DN, WnPk, bn, h_ws, DN, 1024, 16, 8, 256);
    gemm_bf16_f32<<<2048, 256, 0, stream>>>(nfb, DN, WvPk, bv, v_ws, DN, 1024, 16, 8, 256);

    // per-edge attention MLP -> scores + global max
    edge_attn_kernel<<<(int)(N_EDGES / 64), 128, 0, stream>>>(
        nfb, ei, ef, Ppk, Qpk, Rpk, cv, A2pk, a2, s_ws, maxord, N_EDGES);

    softmax_exp_kernel<<<(int)((N_EDGES + 255) / 256), 256, 0, stream>>>(s_ws, maxord, gsum, N_EDGES);

    scatter_kernel<<<(int)(N_EDGES / 4), 256, 0, stream>>>(s_ws, gsum, ei, v_ws, oacc, N_EDGES);

    ln_kernel<<<N_NODES, 256, 0, stream>>>(h_ws, oacc, gamma, beta, out);
}
